// RNN_69020124446839
// MI455X (gfx1250) — matrix operand order repaired and checked by reading, not compiled, not run on hardware
//
#include <hip/hip_runtime.h>
#include <hip/hip_bf16.h>

// ---------------- problem constants ----------------
#define HD   1024      // D_HID
#define DIN  512       // D_IN
#define BB   64        // batch
#define TT   256       // time steps

typedef __attribute__((ext_vector_type(16))) __bf16 v16bf;
typedef __attribute__((ext_vector_type(8)))  __bf16 v8bf;
typedef __attribute__((ext_vector_type(8)))  float  v8f;

__device__ __forceinline__ float sigmoidf_(float x) { return 1.0f / (1.0f + __expf(-x)); }

// CDNA5 async global->LDS copy (ASYNCcnt-tracked). Per the ISA flat-address
// mapping, the low 32 bits of a flat pointer into LDS are the LDS byte address.
__device__ __forceinline__ void async_copy16_to_lds(void* lds_dst, const void* gsrc) {
    unsigned loff = (unsigned)(uintptr_t)lds_dst;
    unsigned long long g = (unsigned long long)(uintptr_t)gsrc;
    asm volatile("global_load_async_to_lds_b128 %0, %1, off"
                 :: "v"(loff), "v"(g) : "memory");
}
__device__ __forceinline__ void async_wait_all() {
    asm volatile("s_wait_asynccnt 0" ::: "memory");
}

// ---------------------------------------------------------------------------
// One-time weight prep: transpose + convert.
//   src [K][N] row-major fp32 -> dst [N][K] (bf16 or f32).
// Makes every WMMA B-fragment two K-contiguous 16B loads per lane.
// ---------------------------------------------------------------------------
__global__ void conv_transpose_bf16(const float* __restrict__ src, __bf16* __restrict__ dst,
                                    int K, int N) {
    size_t idx = (size_t)blockIdx.x * 256 + threadIdx.x;
    size_t total = (size_t)K * N;
    if (idx >= total) return;
    int n = (int)(idx / K);
    int k = (int)(idx % K);
    dst[idx] = (__bf16)src[(size_t)k * N + n];
}

__global__ void conv_transpose_f32(const float* __restrict__ src, float* __restrict__ dst,
                                   int K, int N) {
    size_t idx = (size_t)blockIdx.x * 256 + threadIdx.x;
    size_t total = (size_t)K * N;
    if (idx >= total) return;
    int n = (int)(idx / K);
    int k = (int)(idx % K);
    dst[idx] = src[(size_t)k * N + n];
}

// ---------------------------------------------------------------------------
// One-shot prologue for ALL timesteps (state-independent):
//   xe0all[b][t][k] = bf16(x[b][t][k] + pos_emb0[ra_pos[b][t]][k])
// ---------------------------------------------------------------------------
__global__ void prep_xe0_all(const float* __restrict__ x, const int* __restrict__ ra_pos,
                             const float* __restrict__ pe0, __bf16* __restrict__ xe0all) {
    size_t idx = (size_t)blockIdx.x * 256 + threadIdx.x;
    if (idx >= (size_t)BB * TT * DIN) return;
    int    k  = (int)(idx % DIN);
    size_t bt = idx / DIN;                 // b*T + t
    int pos = ra_pos[bt];
    xe0all[idx] = (__bf16)(x[idx] + pe0[(size_t)pos * DIN + k]);
}

// ---------------------------------------------------------------------------
// Fused gate GEMM + LSTM pointwise + masked state update.
//   z = xe @ W_ih + h @ W_hh + b   (64 x 4096), bf16 WMMA, f32 accumulate.
// Block = one 16-col group x all 4 row tiles (so the 4 row-tile waves share
// B fragments through the WGP$) x 2 K-phases (xe / h), reduced through LDS.
// K loop is double-buffered: next A/B fragments are issued before the four
// WMMAs of the current fragment.
// ---------------------------------------------------------------------------
__global__ __launch_bounds__(256) void gemm_gates_kernel(
    const __bf16* __restrict__ Axe, int Kx, int strideAx,   // [B rows, stride strideAx]
    const __bf16* __restrict__ Ah,                          // [B][H] masked h (prev parity)
    const __bf16* __restrict__ WTx,                         // [4H][Kx] transposed W_ih
    const __bf16* __restrict__ WTh,                         // [4H][H]  transposed W_hh
    const float*  __restrict__ bias,                        // [4H]
    float*        __restrict__ h_state,                     // [B][H] fp32 masked (rw)
    float*        __restrict__ c_state,                     // [B][H] fp32 masked (rw)
    __bf16*       __restrict__ hb_out,                      // [B][H] masked h (next parity)
    float*        __restrict__ h2_buf,                      // [B][H] unmasked h2
    const float*  __restrict__ pad,                         // [B][T]
    int tstep)
{
    __shared__ float redbuf[4][4][256];                     // [rowtile][gate][lane*8+r]

    const int tid  = threadIdx.x;
    const int lane = tid & 31;
    // wave id is uniform across the wave: force it scalar so loop control and
    // base addresses live in SGPRs (no exec-mask loop, SADDR addressing).
    const int wv    = __builtin_amdgcn_readfirstlane(tid >> 5);
    const int phase = wv & 1;                               // 0: xe part, 1: h part
    const int rt    = wv >> 1;                              // 0..3 row tile
    const int cg    = blockIdx.x;                           // 0..63 col group of H
    const int r0 = rt * 16;
    const int n0 = cg * 16;

    const int laneRow = lane & 15;
    const int hiHalf  = lane >> 4;

    const __bf16* A;  const __bf16* WT;  int Kd, strideA;
    if (phase) { A = Ah;  WT = WTh; Kd = HD; strideA = HD; }
    else       { A = Axe; WT = WTx; Kd = Kx; strideA = strideAx; }

    const __bf16* aRow = A + (size_t)(r0 + laneRow) * strideA;
    const __bf16* bRow0 = WT + (size_t)(0 * HD + n0 + laneRow) * Kd;
    const __bf16* bRow1 = WT + (size_t)(1 * HD + n0 + laneRow) * Kd;
    const __bf16* bRow2 = WT + (size_t)(2 * HD + n0 + laneRow) * Kd;
    const __bf16* bRow3 = WT + (size_t)(3 * HD + n0 + laneRow) * Kd;

    // A 16-bit frag: lanes 0-15 hold K [k0,k0+8)+[k0+16,k0+24); lanes 16-31 shift by 8.
    auto loadA = [&](int k0) -> v16bf {
        int ka = k0 + hiHalf * 8;
        v8bf lo = *(const v8bf*)(aRow + ka);
        v8bf hi = *(const v8bf*)(aRow + ka + 16);
        return __builtin_shufflevector(lo, hi, 0,1,2,3,4,5,6,7,8,9,10,11,12,13,14,15);
    };
    // B 16-bit frag: lane = N; lanes 0-15 hold K [k0,k0+8)+[k0+16,k0+24); lanes 16-31 shift by 8.
    auto loadB = [&](const __bf16* brow, int k0) -> v16bf {
        const __bf16* p = brow + k0 + hiHalf * 8;
        v8bf lo = *(const v8bf*)(p);
        v8bf hi = *(const v8bf*)(p + 16);
        return __builtin_shufflevector(lo, hi, 0,1,2,3,4,5,6,7,8,9,10,11,12,13,14,15);
    };

    v8f acc0 = {}, acc1 = {}, acc2 = {}, acc3 = {};

    v16bf aC  = loadA(0);
    v16bf bC0 = loadB(bRow0, 0);
    v16bf bC1 = loadB(bRow1, 0);
    v16bf bC2 = loadB(bRow2, 0);
    v16bf bC3 = loadB(bRow3, 0);

    for (int k0 = 32; k0 < Kd; k0 += 32) {
        // prefetch next fragments before consuming current ones
        v16bf aN  = loadA(k0);
        v16bf bN0 = loadB(bRow0, k0);
        v16bf bN1 = loadB(bRow1, k0);
        v16bf bN2 = loadB(bRow2, k0);
        v16bf bN3 = loadB(bRow3, k0);

        acc0 = __builtin_amdgcn_wmma_f32_16x16x32_bf16(false, aC, false, bC0, (short)0, acc0, false, false);
        acc1 = __builtin_amdgcn_wmma_f32_16x16x32_bf16(false, aC, false, bC1, (short)0, acc1, false, false);
        acc2 = __builtin_amdgcn_wmma_f32_16x16x32_bf16(false, aC, false, bC2, (short)0, acc2, false, false);
        acc3 = __builtin_amdgcn_wmma_f32_16x16x32_bf16(false, aC, false, bC3, (short)0, acc3, false, false);

        aC = aN; bC0 = bN0; bC1 = bN1; bC2 = bN2; bC3 = bN3;
    }
    acc0 = __builtin_amdgcn_wmma_f32_16x16x32_bf16(false, aC, false, bC0, (short)0, acc0, false, false);
    acc1 = __builtin_amdgcn_wmma_f32_16x16x32_bf16(false, aC, false, bC1, (short)0, acc1, false, false);
    acc2 = __builtin_amdgcn_wmma_f32_16x16x32_bf16(false, aC, false, bC2, (short)0, acc2, false, false);
    acc3 = __builtin_amdgcn_wmma_f32_16x16x32_bf16(false, aC, false, bC3, (short)0, acc3, false, false);

    if (phase == 1) {
#pragma unroll
        for (int r = 0; r < 8; ++r) {
            redbuf[rt][0][lane * 8 + r] = acc0[r];
            redbuf[rt][1][lane * 8 + r] = acc1[r];
            redbuf[rt][2][lane * 8 + r] = acc2[r];
            redbuf[rt][3][lane * 8 + r] = acc3[r];
        }
    }
    __syncthreads();

    if (phase == 0) {
        const int col = n0 + laneRow;
#pragma unroll
        for (int r = 0; r < 8; ++r) {
            int row = r0 + r + hiHalf * 8;                   // C/D layout: M = r (+8 hi half)
            float zi = acc0[r] + redbuf[rt][0][lane * 8 + r] + bias[col];
            float zf = acc1[r] + redbuf[rt][1][lane * 8 + r] + bias[HD + col];
            float zg = acc2[r] + redbuf[rt][2][lane * 8 + r] + bias[2 * HD + col];
            float zo = acc3[r] + redbuf[rt][3][lane * 8 + r] + bias[3 * HD + col];

            size_t sidx = (size_t)row * HD + col;
            float co = c_state[sidx];
            float c2 = sigmoidf_(zf) * co + sigmoidf_(zi) * tanhf(zg);
            float h2 = sigmoidf_(zo) * tanhf(c2);

            float m  = pad[row * TT + tstep];
            float om = 1.0f - m;
            float ho = h_state[sidx];
            float hn = ho + om * (h2 - ho);
            float cn = co + om * (c2 - co);

            c_state[sidx] = cn;
            h_state[sidx] = hn;
            hb_out[sidx]  = (__bf16)hn;
            h2_buf[sidx]  = h2;                              // unmasked, for fast-weight kernel
        }
    }
}

// ---------------------------------------------------------------------------
// Fast-weight memory: one workgroup per batch element, F resident in LDS
// (staged with CDNA5 async global->LDS copies).
// ---------------------------------------------------------------------------
__global__ __launch_bounds__(256) void fastweight_kernel(
    const float* __restrict__ h2_buf,     // [B][H] unmasked h2
    float*       __restrict__ F_state,    // [B][16^3]
    const float* __restrict__ WwT,        // [49][H]  transposed W_write
    const float* __restrict__ WrT,        // [32][H]  transposed W_read
    const float* __restrict__ Wo,         // [16][H]
    const float* __restrict__ pad,        // [B][T]
    const int*   __restrict__ ra_pos,     // [B][T]
    const float* __restrict__ pe_next,    // next layer pos_emb [T][H] (or null)
    __bf16*      __restrict__ xe_next,    // next layer bf16 input (or null)
    float*       __restrict__ d_out,      // final output (or null)
    int tstep, int isLast)
{
    __shared__ float sF[4096];
    __shared__ float sh2[HD];
    __shared__ float sw[81];
    __shared__ float sk1[16], sk2[16], sv[16], sdv[16], se[16], sq[16];
    __shared__ float sp[256];
    __shared__ float sbeta;

    const int b    = blockIdx.x;
    const int tid  = threadIdx.x;
    const int lane = tid & 31;
    const int wave = tid >> 5;

    // Stage F (16KB) + h2 (4KB) into LDS with async copies: 16B per lane.
    {
        const float* gF  = F_state + (size_t)b * 4096;
        const float* gh2 = h2_buf  + (size_t)b * HD;
#pragma unroll
        for (int q = 0; q < 4; ++q) {
            int e = (tid + q * 256) * 4;                     // float index, 16B granules
            async_copy16_to_lds(&sF[e], &gF[e]);
        }
        async_copy16_to_lds(&sh2[tid * 4], &gh2[tid * 4]);
        async_wait_all();                                    // per-wave ASYNCcnt
    }
    __syncthreads();                                         // cross-wave visibility

    // 81 dot-products of length H: w[0..48], r[0..31]
    for (int c = wave; c < 81; c += 8) {
        const float* wcol = (c < 49) ? (WwT + (size_t)c * HD) : (WrT + (size_t)(c - 49) * HD);
        float s = 0.0f;
        for (int k = lane; k < HD; k += 32) s += sh2[k] * wcol[k];
        for (int off = 16; off; off >>= 1) s += __shfl_xor(s, off, 32);
        if (lane == 0) sw[c] = s;
    }
    __syncthreads();

    if (tid < 16) {
        sk1[tid] = tanhf(sw[tid]);
        sk2[tid] = tanhf(sw[16 + tid]);
        sv[tid]  = tanhf(sw[32 + tid]);
        se[tid]  = tanhf(sw[49 + tid]);
        sq[tid]  = tanhf(sw[49 + 16 + tid]);
    }
    if (tid == 0) sbeta = sigmoidf_(sw[48]);
    __syncthreads();

    // v_old[k] = sum_ij F[i][j][k] * k1[i] * k2[j]
    {
        int i = tid >> 4, k = tid & 15;
        float s = 0.0f;
#pragma unroll
        for (int j = 0; j < 16; ++j) s += sF[(i * 16 + j) * 16 + k] * sk2[j];
        sp[tid] = s * sk1[i];
    }
    __syncthreads();
    if (tid < 16) {
        float s = 0.0f;
#pragma unroll
        for (int i = 0; i < 16; ++i) s += sp[i * 16 + tid];
        sdv[tid] = sv[tid] - s;          // v - v_old
    }
    __syncthreads();

    const float m    = pad[b * TT + tstep];
    const float om   = 1.0f - m;
    const float beta = sbeta;

    // F2 = F + beta * k1 (x) k2 (x) dv ; global carry gets F + (1-m)*delta
    for (int q = 0; q < 16; ++q) {
        int e = tid + q * 256;
        int i = e >> 8, j = (e >> 4) & 15, k = e & 15;
        float old   = sF[e];
        float delta = beta * sk1[i] * sk2[j] * sdv[k];
        sF[e] = old + delta;                                  // F2 for the reads below
        F_state[(size_t)b * 4096 + e] = old + om * delta;     // masked carry
    }
    __syncthreads();

    // 3 read iterations: e = tanh( sum_ij F2[i][j][k] e[i] q2[j] )
    for (int it = 0; it < 3; ++it) {
        int i = tid >> 4, k = tid & 15;
        float s = 0.0f;
#pragma unroll
        for (int j = 0; j < 16; ++j) s += sF[(i * 16 + j) * 16 + k] * sq[j];
        sp[tid] = s * se[i];
        __syncthreads();
        if (tid < 16) {
            float s2 = 0.0f;
#pragma unroll
            for (int i2 = 0; i2 < 16; ++i2) s2 += sp[i2 * 16 + tid];
            se[tid] = tanhf(s2);
        }
        __syncthreads();
    }

    // out = h2 + tanh(e @ W_o); emit next-layer input or final output
    const int pos = ra_pos[b * TT + tstep];
    for (int n = tid; n < HD; n += 256) {
        float s = 0.0f;
#pragma unroll
        for (int s2 = 0; s2 < 16; ++s2) s += se[s2] * Wo[(size_t)s2 * HD + n];
        float o = sh2[n] + tanhf(s);
        if (isLast) {
            d_out[((size_t)b * TT + tstep) * HD + n] = o * om;
        } else {
            xe_next[(size_t)b * HD + n] = (__bf16)(o + pe_next[(size_t)pos * HD + n]);
        }
    }
}

// ---------------------------------------------------------------------------
// Host side
// ---------------------------------------------------------------------------
extern "C" void kernel_launch(void* const* d_in, const int* in_sizes, int n_in,
                              void* d_out, int out_size, void* d_ws, size_t ws_size,
                              hipStream_t stream) {
    (void)n_in; (void)out_size; (void)ws_size;

    const float* x      = (const float*)d_in[0];
    const int*   ra_pos = (const int*)  d_in[2];
    const float* pad    = (const float*)d_in[3];

    // Param leaves: detect JAX sorted-key vs insertion-order flattening by size.
    // sorted:    W_hh, W_ih, W_o, W_read, W_write, b, pos_emb
    // insertion: W_ih, W_hh, b, W_write, W_read, W_o, pos_emb
    const bool sortedOrder = (in_sizes[4] == HD * 4 * HD);   // W_hh0 first => 4,194,304
    const float *W_ih[2], *W_hh[2], *W_o[2], *W_rd[2], *W_wr[2], *bias[2], *pe[2];
    for (int l = 0; l < 2; ++l) {
        const int base = 4 + 7 * l;
        if (sortedOrder) {
            W_hh[l] = (const float*)d_in[base + 0];
            W_ih[l] = (const float*)d_in[base + 1];
            W_o [l] = (const float*)d_in[base + 2];
            W_rd[l] = (const float*)d_in[base + 3];
            W_wr[l] = (const float*)d_in[base + 4];
            bias[l] = (const float*)d_in[base + 5];
            pe  [l] = (const float*)d_in[base + 6];
        } else {
            W_ih[l] = (const float*)d_in[base + 0];
            W_hh[l] = (const float*)d_in[base + 1];
            bias[l] = (const float*)d_in[base + 2];
            W_wr[l] = (const float*)d_in[base + 3];
            W_rd[l] = (const float*)d_in[base + 4];
            W_o [l] = (const float*)d_in[base + 5];
            pe  [l] = (const float*)d_in[base + 6];
        }
    }
    const int Kin[2] = { DIN, HD };

    // ---- workspace carve-out (256B aligned) ----
    char* ws = (char*)d_ws;
    size_t off = 0;
    auto alloc = [&](size_t bytes) -> char* {
        char* p = ws + off;
        off = (off + bytes + 255) & ~(size_t)255;
        return p;
    };

    __bf16* WT_ih[2]; __bf16* WT_hh[2];
    float*  WwT[2];   float*  WrT[2];
    WT_ih[0] = (__bf16*)alloc((size_t)4 * HD * DIN * 2);
    WT_hh[0] = (__bf16*)alloc((size_t)4 * HD * HD  * 2);
    WT_ih[1] = (__bf16*)alloc((size_t)4 * HD * HD  * 2);
    WT_hh[1] = (__bf16*)alloc((size_t)4 * HD * HD  * 2);
    for (int l = 0; l < 2; ++l) {
        WwT[l] = (float*)alloc((size_t)49 * HD * 4);
        WrT[l] = (float*)alloc((size_t)32 * HD * 4);
    }
    float* h_st[2]; float* c_st[2]; float* F_st[2]; float* h2b[2];
    __bf16* hb[2][2];
    for (int l = 0; l < 2; ++l) {
        h_st[l] = (float*)alloc((size_t)BB * HD * 4);
        c_st[l] = (float*)alloc((size_t)BB * HD * 4);
        F_st[l] = (float*)alloc((size_t)BB * 4096 * 4);
        h2b[l]  = (float*)alloc((size_t)BB * HD * 4);
        hb[l][0] = (__bf16*)alloc((size_t)BB * HD * 2);
        hb[l][1] = (__bf16*)alloc((size_t)BB * HD * 2);
    }
    __bf16* xe0all = (__bf16*)alloc((size_t)BB * TT * DIN * 2);   // all timesteps
    __bf16* xe1    = (__bf16*)alloc((size_t)BB * HD * 2);

    // ---- zero recurrent state (graph-capturable memsets) ----
    for (int l = 0; l < 2; ++l) {
        hipMemsetAsync(h_st[l], 0, (size_t)BB * HD * 4, stream);
        hipMemsetAsync(c_st[l], 0, (size_t)BB * HD * 4, stream);
        hipMemsetAsync(F_st[l], 0, (size_t)BB * 4096 * 4, stream);
        hipMemsetAsync(hb[l][0], 0, (size_t)BB * HD * 2, stream);
        hipMemsetAsync(hb[l][1], 0, (size_t)BB * HD * 2, stream);
    }

    // ---- one-time weight transpose/convert + full xe0 precompute ----
    for (int l = 0; l < 2; ++l) {
        size_t tih = (size_t)Kin[l] * 4 * HD;
        conv_transpose_bf16<<<(int)((tih + 255) / 256), 256, 0, stream>>>(W_ih[l], WT_ih[l], Kin[l], 4 * HD);
        size_t thh = (size_t)HD * 4 * HD;
        conv_transpose_bf16<<<(int)((thh + 255) / 256), 256, 0, stream>>>(W_hh[l], WT_hh[l], HD, 4 * HD);
        conv_transpose_f32<<<(int)(((size_t)HD * 49 + 255) / 256), 256, 0, stream>>>(W_wr[l], WwT[l], HD, 49);
        conv_transpose_f32<<<(int)(((size_t)HD * 32 + 255) / 256), 256, 0, stream>>>(W_rd[l], WrT[l], HD, 32);
    }
    {
        size_t tot = (size_t)BB * TT * DIN;
        prep_xe0_all<<<(int)((tot + 255) / 256), 256, 0, stream>>>(x, ra_pos, pe[0], xe0all);
    }

    // ---- time recurrence: 4 launches per step ----
    for (int t = 0; t < TT; ++t) {
        const int p = t & 1;

        gemm_gates_kernel<<<64, 256, 0, stream>>>(
            xe0all + (size_t)t * DIN, DIN, TT * DIN,          // row b at xe0all[(b*T+t)*DIN]
            hb[0][p], WT_ih[0], WT_hh[0], bias[0],
            h_st[0], c_st[0], hb[0][p ^ 1], h2b[0], pad, t);

        fastweight_kernel<<<BB, 256, 0, stream>>>(
            h2b[0], F_st[0], WwT[0], WrT[0], W_o[0], pad, ra_pos,
            pe[1], xe1, nullptr, t, 0);

        gemm_gates_kernel<<<64, 256, 0, stream>>>(
            xe1, HD, HD,
            hb[1][p], WT_ih[1], WT_hh[1], bias[1],
            h_st[1], c_st[1], hb[1][p ^ 1], h2b[1], pad, t);

        fastweight_kernel<<<BB, 256, 0, stream>>>(
            h2b[1], F_st[1], WwT[1], WrT[1], W_o[1], pad, ra_pos,
            nullptr, nullptr, (float*)d_out, t, 1);
    }
}
